// EX_Module_noself_7670811590903
// MI455X (gfx1250) — matrix operand order repaired and checked by reading, not compiled, not run on hardware
//
#include <hip/hip_runtime.h>

// Problem constants
#define B_    16
#define C_    1024
#define CH_   512
#define HW_   4096
#define KDIM  1024
#define MROWS 1664   // 512(vr) + 512(ql) + 512(vl) + 1(qr) padded to 13*128

typedef __attribute__((ext_vector_type(16))) __bf16 v16bf;
typedef __attribute__((ext_vector_type(8)))  float  v8f;

union FragBF { uint4 u[2]; v16bf v; };

#if __has_builtin(__builtin_amdgcn_tensor_load_to_lds)
#define USE_TDM 1
typedef __attribute__((ext_vector_type(4))) unsigned int u32x4_t;
typedef __attribute__((ext_vector_type(8))) int          i32x8_t;
typedef __attribute__((ext_vector_type(4))) int          i32x4_t;
#else
#define USE_TDM 0
#endif

__device__ __forceinline__ unsigned short f2bf(float f) {
  unsigned int u = __float_as_uint(f);
  u += 0x7FFFu + ((u >> 16) & 1u);     // round-to-nearest-even
  return (unsigned short)(u >> 16);
}
__device__ __forceinline__ float bf2f(unsigned short h) {
  return __uint_as_float(((unsigned int)h) << 16);
}
__device__ __forceinline__ float sigmoidf_(float x) { return 1.0f / (1.0f + __expf(-x)); }

// ---------------------------------------------------------------------------
// Kernel 1: stack + convert the four conv weights into one bf16 matrix
//   rows [0,512) = w_vr, [512,1024) = w_ql, [1024,1536) = w_vl, 1536 = w_qr,
//   [1537,1664) = zero padding.
// ---------------------------------------------------------------------------
__global__ void pack_weights_kernel(const float* __restrict__ wqr,
                                    const float* __restrict__ wvr,
                                    const float* __restrict__ wql,
                                    const float* __restrict__ wvl,
                                    unsigned short* __restrict__ Wbf) {
  int idx = blockIdx.x * 256 + threadIdx.x;     // MROWS*KDIM total
  int row = idx >> 10, col = idx & 1023;
  float v = 0.f;
  if      (row < 512)   v = wvr[row * KDIM + col];
  else if (row < 1024)  v = wql[(row - 512) * KDIM + col];
  else if (row < 1536)  v = wvl[(row - 1024) * KDIM + col];
  else if (row == 1536) v = wqr[col];
  Wbf[idx] = f2bf(v);
}

// ---------------------------------------------------------------------------
// Kernel 2: fused GEMM  Y[o,n] = relu( sum_c W[o,c] * x[b,c,n] )  via bf16 WMMA.
// D tile: M = n (spatial), N = o (output channel).
// Block = 256 threads (8 waves), tile 128n x 128o, BK = 64.
// Wave grid: 4 (n) x 2 (o) -> each wave 32n x 64o = 2x4 WMMA accumulators.
// B tile (weights, already bf16) is fetched by the Tensor Data Mover:
// one tensor_load_to_lds per K-stage, issued by wave 0, overlapping the
// VALU f32->bf16 convert+transpose of the A tile.
// ---------------------------------------------------------------------------
__global__ __launch_bounds__(256)
void gemm_attn_kernel(const float* __restrict__ x,
                      const unsigned short* __restrict__ Wbf,
                      unsigned short* __restrict__ Yvr,
                      unsigned short* __restrict__ Yvl,
                      float* __restrict__ ql_sum,
                      float* __restrict__ qr_out) {
  const int nb   = blockIdx.x;          // 0..31  (n tiles of 128)
  const int ob   = blockIdx.y;          // 0..12  (o tiles of 128)
  const int b    = blockIdx.z;          // 0..15
  const int tid  = threadIdx.x;
  const int lane = tid & 31;
  const int wave = tid >> 5;
  const int wn   = wave & 3;            // n sub-tile (32 rows)
  const int wo   = wave >> 2;           // o sub-tile (64 cols)
  const int l15  = lane & 15;
  const int half = lane >> 4;

  __shared__ __align__(64) unsigned short sA[128 * 64];  // x^T tile: [n][k], stride 64
  __shared__ __align__(64) unsigned short sB[128 * 64];  // W   tile: [o][k], stride 64

  v8f acc[2][4];
  #pragma unroll
  for (int m = 0; m < 2; ++m)
    #pragma unroll
    for (int j = 0; j < 4; ++j)
      #pragma unroll
      for (int r = 0; r < 8; ++r) acc[m][j][r] = 0.f;

  const float*          xb = x   + (size_t)b * C_ * HW_ + nb * 128;
  const unsigned short* wb = Wbf + (size_t)ob * 128 * KDIM;

  for (int k0 = 0; k0 < KDIM; k0 += 64) {
#if USE_TDM
    // ---- B tile via Tensor Data Mover: 2D tile (64 k-contig x 128 o-rows)
    // of the (MROWS x KDIM) bf16 tensor -> LDS rows packed back-to-back,
    // i.e. exactly the sB[o][k] stride-64 layout the fragments read.
    if (wave == 0) {
      unsigned long long ga =
          (unsigned long long)(size_t)wb + (unsigned long long)k0 * 2ull;
      u32x4_t g0;
      g0[0] = 1u;                                           // count=1, user desc
      g0[1] = (unsigned int)(size_t)(void*)sB;              // lds_addr (bytes)
      g0[2] = (unsigned int)ga;                             // global_addr[31:0]
      g0[3] = ((unsigned int)(ga >> 32) & 0x1FFFFFFu)       // global_addr[56:32]
              | (2u << 30);                                 // type=2 ("image")
      i32x8_t g1;
      g1[0] = (int)(1u << 16);                              // data_size=1 -> 2 bytes
      g1[1] = (int)((KDIM & 0xFFFFu) << 16);                // tensor_dim0[15:0]
      g1[2] = (int)((KDIM >> 16) | ((MROWS & 0xFFFFu) << 16)); // td0[31:16]|td1[15:0]
      g1[3] = (int)((MROWS >> 16) | (64u << 16));           // td1[31:16]|tile_dim0=64
      g1[4] = (int)128u;                                    // tile_dim1=128, tile_dim2=0
      g1[5] = (int)KDIM;                                    // tensor_dim0_stride[31:0]
      g1[6] = (int)((KDIM & 0xFFFFu) << 16);                // s0[47:32]=0 | s1[15:0]
      g1[7] = 0;                                            // tensor_dim1_stride[47:16]
      i32x4_t gz4 = {0, 0, 0, 0};                           // groups 2/3 unused (2D)
      i32x8_t gz8 = {0, 0, 0, 0, 0, 0, 0, 0};               // trailing group (zero-fill)
      __builtin_amdgcn_tensor_load_to_lds(g0, g1, gz4, gz4, gz8, 0);
    }
#else
    // ---- fallback: B tile via VGPR round-trip copy
    #pragma unroll
    for (int i = 0; i < 4; ++i) {
      int q = tid + i * 256;            // 0..1023 16B chunks
      int row = q >> 3, c8 = q & 7;
      *reinterpret_cast<uint4*>(&sB[row * 64 + c8 * 8]) =
          *reinterpret_cast<const uint4*>(wb + (size_t)row * KDIM + k0 + c8 * 8);
    }
#endif

    // ---- global -> LDS: A = x^T (128n x 64k), f32 load + bf16 convert + transpose
    #pragma unroll
    for (int i = 0; i < 8; ++i) {
      int q  = tid + i * 256;           // 0..2047 float4 chunks
      int cc = q >> 5;                  // k row 0..63
      int n4 = q & 31;                  // float4 index along n
      float4 xv = *reinterpret_cast<const float4*>(xb + (size_t)(k0 + cc) * HW_ + n4 * 4);
      int nb0 = n4 * 4;
      sA[(nb0 + 0) * 64 + cc] = f2bf(xv.x);
      sA[(nb0 + 1) * 64 + cc] = f2bf(xv.y);
      sA[(nb0 + 2) * 64 + cc] = f2bf(xv.z);
      sA[(nb0 + 3) * 64 + cc] = f2bf(xv.w);
    }

#if USE_TDM
    if (wave == 0) __builtin_amdgcn_s_wait_tensorcnt(0);    // DMA landed in LDS
#endif
    __syncthreads();

    #pragma unroll
    for (int kk = 0; kk < 64; kk += 32) {
      FragBF afr[2], bfr[4];
      // A fragment (16x32 bf16): lanes 0-15 row M=l hold K 0-7 & 16-23,
      // lanes 16-31 row M=l-16 hold K 8-15 & 24-31.
      #pragma unroll
      for (int m = 0; m < 2; ++m) {
        int row   = wn * 32 + m * 16 + l15;
        int cbase = kk + half * 8;
        afr[m].u[0] = *reinterpret_cast<const uint4*>(&sA[row * 64 + cbase]);
        afr[m].u[1] = *reinterpret_cast<const uint4*>(&sA[row * 64 + cbase + 16]);
      }
      // B fragment (32x16 bf16): lane = column o; lanes 0-15 hold K 0-7 & 16-23,
      // lanes 16-31 hold K 8-15 & 24-31 (same K order as A).
      #pragma unroll
      for (int j = 0; j < 4; ++j) {
        int row = wo * 64 + j * 16 + l15;
        bfr[j].u[0] = *reinterpret_cast<const uint4*>(&sB[row * 64 + kk + half * 8]);
        bfr[j].u[1] = *reinterpret_cast<const uint4*>(&sB[row * 64 + kk + half * 8 + 16]);
      }
      #pragma unroll
      for (int m = 0; m < 2; ++m)
        #pragma unroll
        for (int j = 0; j < 4; ++j)
          acc[m][j] = __builtin_amdgcn_wmma_f32_16x16x32_bf16(
              false, afr[m].v, false, bfr[j].v, (short)0, acc[m][j], false, false);
    }
    __syncthreads();
  }

  // ---- epilogue: ReLU + region-routed output
  // C/D layout: value r of lane -> (M = r + 8*half, N = l15)
  #pragma unroll
  for (int m = 0; m < 2; ++m) {
    #pragma unroll
    for (int j = 0; j < 4; ++j) {
      int o_g  = ob * 128 + wo * 64 + j * 16 + l15;
      int n_g0 = nb * 128 + wn * 32 + m * 16 + half * 8;
      float v[8];
      #pragma unroll
      for (int r = 0; r < 8; ++r) v[r] = fmaxf(acc[m][j][r], 0.f);

      if (o_g < 512) {                                   // input_x (vr branch)
        unsigned short* p = Yvr + ((size_t)(b * 512 + o_g)) * HW_ + n_g0;
        #pragma unroll
        for (int r = 0; r < 8; r += 2) {
          unsigned int pk = (unsigned int)f2bf(v[r]) | ((unsigned int)f2bf(v[r + 1]) << 16);
          *reinterpret_cast<unsigned int*>(p + r) = pk;
        }
      } else if (o_g < 1024) {                           // g_x (ql): reduce over n only
        float s = v[0] + v[1] + v[2] + v[3] + v[4] + v[5] + v[6] + v[7];
        atomicAdd(&ql_sum[b * 512 + (o_g - 512)], s);
      } else if (o_g < 1536) {                           // theta_x (vl branch)
        unsigned short* p = Yvl + ((size_t)(b * 512 + (o_g - 1024))) * HW_ + n_g0;
        #pragma unroll
        for (int r = 0; r < 8; r += 2) {
          unsigned int pk = (unsigned int)f2bf(v[r]) | ((unsigned int)f2bf(v[r + 1]) << 16);
          *reinterpret_cast<unsigned int*>(p + r) = pk;
        }
      } else if (o_g == 1536) {                          // qr row (pre-softmax logits)
        float* p = qr_out + b * HW_ + n_g0;
        #pragma unroll
        for (int r = 0; r < 8; ++r) p[r] = v[r];
      }
    }
  }
}

// ---------------------------------------------------------------------------
// Kernel 3 (one block per batch): qr softmax -> context -> layernorm+sigmoid
// (spatial_attn), plus softmax of ql mean (avg_x).
// ---------------------------------------------------------------------------
__global__ __launch_bounds__(256)
void attn_stats_kernel(const float* __restrict__ qr_out,
                       const unsigned short* __restrict__ Yvr,
                       const float* __restrict__ ql_sum,
                       float* __restrict__ spatial,
                       float* __restrict__ avg_x) {
  const int b = blockIdx.x, t = threadIdx.x;
  __shared__ float red[256];
  __shared__ float mask[HW_];
  __shared__ float ctx[CH_];

  // --- softmax over qr logits (4096) ---
  float mx = -1e30f;
  for (int i = t; i < HW_; i += 256) mx = fmaxf(mx, qr_out[b * HW_ + i]);
  red[t] = mx; __syncthreads();
  for (int s = 128; s > 0; s >>= 1) { if (t < s) red[t] = fmaxf(red[t], red[t + s]); __syncthreads(); }
  mx = red[0]; __syncthreads();
  float sm = 0.f;
  for (int i = t; i < HW_; i += 256) { float e = __expf(qr_out[b * HW_ + i] - mx); mask[i] = e; sm += e; }
  red[t] = sm; __syncthreads();
  for (int s = 128; s > 0; s >>= 1) { if (t < s) red[t] += red[t + s]; __syncthreads(); }
  float inv = 1.f / red[0]; __syncthreads();
  for (int i = t; i < HW_; i += 256) mask[i] *= inv;
  __syncthreads();

  // --- context[c] = sum_n Yvr[b,c,n] * mask[n] ---
  for (int c = t; c < CH_; c += 256) {
    const unsigned short* row = Yvr + ((size_t)(b * CH_ + c)) * HW_;
    float a = 0.f;
    for (int n = 0; n < HW_; ++n) a += bf2f(row[n]) * mask[n];
    ctx[c] = a;
  }
  __syncthreads();

  // --- layernorm over 512 channels + sigmoid ---
  float s1 = 0.f, s2 = 0.f;
  for (int c = t; c < CH_; c += 256) { float v = ctx[c]; s1 += v; s2 += v * v; }
  red[t] = s1; __syncthreads();
  for (int s = 128; s > 0; s >>= 1) { if (t < s) red[t] += red[t + s]; __syncthreads(); }
  float mu = red[0] / (float)CH_; __syncthreads();
  red[t] = s2; __syncthreads();
  for (int s = 128; s > 0; s >>= 1) { if (t < s) red[t] += red[t + s]; __syncthreads(); }
  float var = red[0] / (float)CH_ - mu * mu; __syncthreads();
  float rstd = rsqrtf(var + 1e-5f);
  for (int c = t; c < CH_; c += 256)
    spatial[b * CH_ + c] = sigmoidf_((ctx[c] - mu) * rstd);

  // --- avg_x = softmax over channels of mean(g_x) ---
  float vmx = -1e30f;
  for (int c = t; c < CH_; c += 256) vmx = fmaxf(vmx, ql_sum[b * CH_ + c] * (1.f / HW_));
  red[t] = vmx; __syncthreads();
  for (int s = 128; s > 0; s >>= 1) { if (t < s) red[t] = fmaxf(red[t], red[t + s]); __syncthreads(); }
  vmx = red[0]; __syncthreads();
  float vsum = 0.f;
  for (int c = t; c < CH_; c += 256) vsum += __expf(ql_sum[b * CH_ + c] * (1.f / HW_) - vmx);
  red[t] = vsum; __syncthreads();
  for (int s = 128; s > 0; s >>= 1) { if (t < s) red[t] += red[t + s]; __syncthreads(); }
  float vinv = 1.f / red[0]; __syncthreads();
  for (int c = t; c < CH_; c += 256)
    avg_x[b * CH_ + c] = __expf(ql_sum[b * CH_ + c] * (1.f / HW_) - vmx) * vinv;
}

// ---------------------------------------------------------------------------
// Kernel 4: channel_attn[b,n] = sigmoid( sum_c avg_x[b,c] * theta_x[b,c,n] )
// ---------------------------------------------------------------------------
__global__ __launch_bounds__(256)
void channel_attn_kernel(const unsigned short* __restrict__ Yvl,
                         const float* __restrict__ avg_x,
                         float* __restrict__ channel) {
  const int b  = blockIdx.x >> 4;
  const int nb = blockIdx.x & 15;
  const int t  = threadIdx.x;
  __shared__ float sav[CH_];
  sav[t]       = avg_x[b * CH_ + t];
  sav[t + 256] = avg_x[b * CH_ + t + 256];
  __syncthreads();
  const int n = nb * 256 + t;
  const unsigned short* base = Yvl + (size_t)b * CH_ * HW_ + n;
  float a = 0.f;
  #pragma unroll 4
  for (int c = 0; c < CH_; ++c) a += sav[c] * bf2f(base[(size_t)c * HW_]);
  channel[b * HW_ + n] = sigmoidf_(a);
}

// ---------------------------------------------------------------------------
// Kernel 5: out = x * (1 + sk_attn);  c<512: sp*chn (sequence), else sp+chn.
// ---------------------------------------------------------------------------
__global__ __launch_bounds__(256)
void apply_attn_kernel(const float* __restrict__ x,
                       const float* __restrict__ spatial,
                       const float* __restrict__ channel,
                       float* __restrict__ out) {
  size_t idx = (size_t)blockIdx.x * 256 + threadIdx.x;   // float4 index
  size_t e = idx * 4;
  int n = (int)(e & (HW_ - 1));
  int c = (int)((e >> 12) & (C_ - 1));
  int b = (int)(e >> 22);
  float4 xv = reinterpret_cast<const float4*>(x)[idx];
  float4 cv = reinterpret_cast<const float4*>(channel)[(b * HW_ + n) >> 2];
  float sp = spatial[b * CH_ + (c & (CH_ - 1))];
  float4 o;
  if (c < CH_) {
    o.x = xv.x * (1.f + sp * cv.x);
    o.y = xv.y * (1.f + sp * cv.y);
    o.z = xv.z * (1.f + sp * cv.z);
    o.w = xv.w * (1.f + sp * cv.w);
  } else {
    o.x = xv.x * (1.f + sp + cv.x);
    o.y = xv.y * (1.f + sp + cv.y);
    o.z = xv.z * (1.f + sp + cv.z);
    o.w = xv.w * (1.f + sp + cv.w);
  }
  reinterpret_cast<float4*>(out)[idx] = o;
}

// ---------------------------------------------------------------------------
extern "C" void kernel_launch(void* const* d_in, const int* in_sizes, int n_in,
                              void* d_out, int out_size, void* d_ws, size_t ws_size,
                              hipStream_t stream) {
  const float* x   = (const float*)d_in[0];
  const float* wqr = (const float*)d_in[1];
  const float* wvr = (const float*)d_in[2];
  const float* wql = (const float*)d_in[3];
  const float* wvl = (const float*)d_in[4];

  char* ws = (char*)d_ws;
  // workspace layout (bytes)
  unsigned short* Wbf   = (unsigned short*)(ws + 0);          //  3,407,872
  unsigned short* Yvr   = (unsigned short*)(ws + 3407872);    // 67,108,864
  unsigned short* Yvl   = (unsigned short*)(ws + 70516736);   // 67,108,864
  float*          ql_s  = (float*)(ws + 137625600);           //     32,768
  float*          qr_o  = (float*)(ws + 137658368);           //    262,144
  float*          spat  = (float*)(ws + 137920512);           //     32,768
  float*          avgx  = (float*)(ws + 137953280);           //     32,768
  float*          chn   = (float*)(ws + 137986048);           //    262,144

  (void)hipMemsetAsync(ql_s, 0, (size_t)B_ * CH_ * sizeof(float), stream);

  pack_weights_kernel<<<MROWS * KDIM / 256, 256, 0, stream>>>(wqr, wvr, wql, wvl, Wbf);

  dim3 g(HW_ / 128, MROWS / 128, B_);   // (32, 13, 16)
  gemm_attn_kernel<<<g, 256, 0, stream>>>(x, Wbf, Yvr, Yvl, ql_s, qr_o);

  attn_stats_kernel<<<B_, 256, 0, stream>>>(qr_o, Yvr, ql_s, spat, avgx);
  channel_attn_kernel<<<B_ * 16, 256, 0, stream>>>(Yvl, avgx, chn);

  size_t total4 = (size_t)B_ * C_ * HW_ / 4;   // 16,777,216 float4s
  apply_attn_kernel<<<(unsigned)(total4 / 256), 256, 0, stream>>>(x, spat, chn, (float*)d_out);
}
